// MlpwithSOMModule_8151847928473
// MI455X (gfx1250) — matrix operand order repaired and checked by reading, not compiled, not run on hardware
//
#include <hip/hip_runtime.h>

// ---------------------------------------------------------------------------
// Problem constants (from reference): B=16, K=64, L=128, D=768
// ---------------------------------------------------------------------------
constexpr int BK   = 16 * 64;   // 1024 (b,k) pairs -> grid size
constexpr int LL   = 128;       // rows per pair
constexpr int DD   = 768;       // feature dim
constexpr int KCH  = 64;        // K-chunk staged in LDS per step
constexpr int XS   = KCH + 8;   // LDS row stride (bf16 elems) for staged chunks
constexpr int SIMS = LL + 4;    // LDS row stride (f32) for sim matrix

// Kernel 1 LDS: ctx chunk + ent chunk (bf16) + sim (f32)
constexpr int SMEM1 = 2 * LL * XS * 2 + LL * SIMS * 4;   // 104448 B

// Kernel 2 LDS layout (bytes). Region A holds h0/h2/h4, region B holds
// x-chunk (layer0) then h1/h3/h5. Total 318976 B < 320 KB per WGP.
constexpr int S0 = 776, S1 = 392, S2 = 200, S3 = 104, S4 = 72, S5 = 72;
constexpr int OFF_B   = LL * S0 * 2;            // 198656
constexpr int OFF_W   = OFF_B + LL * S1 * 2;    // 299008
constexpr int OFF_IDX = OFF_W + LL * XS * 2;    // 317440
constexpr int OFF_RED = OFF_IDX + LL * 4;       // 317952
constexpr int SMEM2   = OFF_RED + 256 * 4;      // 318976

// Scheduling barrier: pin the fragment-load burst before the WMMA burst so the
// scheduler cannot sink loads to their uses (which re-serializes ds<->wmma).
#if defined(__has_builtin)
#  if __has_builtin(__builtin_amdgcn_sched_barrier)
#    define SCHED_FENCE() __builtin_amdgcn_sched_barrier(0)
#  endif
#endif
#ifndef SCHED_FENCE
#  define SCHED_FENCE()
#endif

// ---------------------------------------------------------------------------
// WMMA plumbing (CDNA5 wave32, v_wmma_f32_16x16x32_bf16)
// ---------------------------------------------------------------------------
typedef __attribute__((ext_vector_type(16))) __bf16 v16bf;
typedef __attribute__((ext_vector_type(8)))  float  v8f;
typedef __attribute__((ext_vector_type(4)))  unsigned int u32x4;
typedef __attribute__((ext_vector_type(2)))  unsigned int u32x2;

union Frag16 {
    u32x4 q[2];
    v16bf v;
};

__device__ __forceinline__ unsigned short f2bf(float f) {
    unsigned int x = __float_as_uint(f);
    unsigned int r = x + 0x7FFFu + ((x >> 16) & 1u);   // round-to-nearest-even
    return (unsigned short)(r >> 16);
}
__device__ __forceinline__ float bf2f(unsigned short h) {
    return __uint_as_float(((unsigned int)h) << 16);
}
__device__ __forceinline__ void pack4(unsigned short* dst, float4 v) {
    u32x2 u;
    u.x = (unsigned int)f2bf(v.x) | ((unsigned int)f2bf(v.y) << 16);
    u.y = (unsigned int)f2bf(v.z) | ((unsigned int)f2bf(v.w) << 16);
    *(u32x2*)dst = u;   // 8B-aligned by construction
}

// A-matrix 16x32 bf16 fragment (ISA 7.12.2): lane m=lane&15, half h=lane>>4,
// elements 0..7 -> K = kbase + h*8 .. +7 ; elements 8..15 -> K = kbase+16+h*8.
__device__ __forceinline__ v16bf load_frag_a(const unsigned short* lds, int stride,
                                             int row0, int kbase, int lane) {
    const int m = lane & 15, h = lane >> 4;
    const unsigned short* p = lds + (row0 + m) * stride + kbase + h * 8;
    Frag16 f;
    f.q[0] = *(const u32x4*)(p);        // 16B ds_load
    f.q[1] = *(const u32x4*)(p + 16);   // 16B ds_load
    return f.v;
}
// B-matrix 32x16 bf16 fragment: lane n=lane&15 selects N (row of W / ent),
// half selects K group: elements 0..7 -> K = kbase + h*8 .. +7 ; 8..15 -> kbase+16+h*8.
__device__ __forceinline__ v16bf load_frag_b(const unsigned short* lds, int stride,
                                             int row0, int kbase, int lane) {
    const int n = lane & 15, h = lane >> 4;
    const unsigned short* p = lds + (row0 + n) * stride + kbase + h * 8;
    Frag16 f;
    f.q[0] = *(const u32x4*)(p);
    f.q[1] = *(const u32x4*)(p + 16);
    return f.v;
}

__device__ __forceinline__ v8f wmma_bf16(v16bf a, v16bf b, v8f c) {
    // 8 args: (neg_a, A, neg_b, B, c_mod, C, reuse_a, reuse_b)
    return __builtin_amdgcn_wmma_f32_16x16x32_bf16(false, a, false, b,
                                                   (short)0, c, false, false);
}

// ---------------------------------------------------------------------------
// Kernel 1: sim = ctx @ ent^T (128x128, K=768), row argmax -> idx[bk][l]
// One workgroup per (b,k); 8 waves; wave w owns output tile-row w.
// ---------------------------------------------------------------------------
__global__ __launch_bounds__(256)
void som_sim_argmax(const float* __restrict__ context, int* __restrict__ idx_out) {
    extern __shared__ char smem[];
    unsigned short* sctx = (unsigned short*)smem;             // [128][XS] bf16
    unsigned short* sent = sctx + LL * XS;                    // [128][XS] bf16
    float*          ssim = (float*)(sent + LL * XS);          // [128][SIMS] f32

    const int bk   = blockIdx.x;
    const int tid  = threadIdx.x;
    const int wave = tid >> 5;
    const int lane = tid & 31;

    const float* ctx = context + (size_t)bk * 2 * LL * DD;   // context[b,k,0]
    const float* ent = ctx + (size_t)LL * DD;                // context[b,k,1]

    v8f acc[8];
#pragma unroll
    for (int t = 0; t < 8; ++t)
#pragma unroll
        for (int r = 0; r < 8; ++r) acc[t][r] = 0.0f;

    for (int dc = 0; dc < DD / KCH; ++dc) {
        __syncthreads();
        // Stage 128x64 fp32 -> bf16 for both ctx and ent (2 threads per row).
        {
            const int row = tid >> 1;
            const int c0  = (tid & 1) * 32;
            const float* gc = ctx + (size_t)row * DD + dc * KCH + c0;
            const float* ge = ent + (size_t)row * DD + dc * KCH + c0;
            unsigned short* lc = sctx + row * XS + c0;
            unsigned short* le = sent + row * XS + c0;
#pragma unroll
            for (int j = 0; j < 32; j += 4) {
                pack4(lc + j, *(const float4*)(gc + j));
                pack4(le + j, *(const float4*)(ge + j));
            }
            if (dc + 1 < DD / KCH) {                // gfx1250 global_prefetch_b8
                __builtin_prefetch(gc + KCH, 0, 0);
                __builtin_prefetch(ge + KCH, 0, 0);
            }
        }
        __syncthreads();
#pragma unroll
        for (int ks = 0; ks < KCH / 32; ++ks) {
            // Burst all fragment loads, fence the scheduler, then fire 8
            // independent WMMAs: one dscnt wait per k-step instead of 8.
            v16bf a = load_frag_a(sctx, XS, wave * 16, ks * 32, lane);
            v16bf bfr[8];
#pragma unroll
            for (int tn = 0; tn < 8; ++tn)
                bfr[tn] = load_frag_b(sent, XS, tn * 16, ks * 32, lane);
            SCHED_FENCE();
#pragma unroll
            for (int tn = 0; tn < 8; ++tn)
                acc[tn] = wmma_bf16(a, bfr[tn], acc[tn]);
            SCHED_FENCE();
        }
    }

    // Dump accumulators to LDS. C layout: VGPR r -> M=r (lanes 0-15) / M=r+8.
#pragma unroll
    for (int tn = 0; tn < 8; ++tn) {
        const int n     = tn * 16 + (lane & 15);
        const int rbase = wave * 16 + (lane >> 4) * 8;
#pragma unroll
        for (int r = 0; r < 8; ++r)
            ssim[(rbase + r) * SIMS + n] = acc[tn][r];
    }
    __syncthreads();

    // Row argmax (first occurrence of max, like jnp.argmax).
    if (tid < LL) {
        const float* row = ssim + tid * SIMS;
        float best = row[0];
        int   bi   = 0;
        for (int j = 1; j < LL; ++j) {
            float v = row[j];
            if (v > best) { best = v; bi = j; }
        }
        idx_out[(size_t)bk * LL + tid] = bi;
    }
}

// ---------------------------------------------------------------------------
// Kernel 2: fused MLP. One workgroup per (b,k), activations live in LDS bf16.
// ---------------------------------------------------------------------------
struct MlpParams {
    const float* ctx_base;
    const int*   idx;
    const float* W[7];
    const float* Bv[7];
    float*       out;
};

// Generic WMMA layer: out[128][Dout] = tanh(in[128][Din] @ W^T + b), bf16 LDS.
template<int Din, int Dout>
__device__ __forceinline__ void mlp_layer(const unsigned short* __restrict__ in_lds, int in_stride,
                                          unsigned short* __restrict__ out_lds, int out_stride,
                                          unsigned short* __restrict__ wch,
                                          const float* __restrict__ Wg,
                                          const float* __restrict__ bg, int tid) {
    const int wave = tid >> 5;
    const int lane = tid & 31;
    constexpr int NOC = (Dout + 127) / 128;
    constexpr int NKC = (Din + KCH - 1) / KCH;

#pragma unroll
    for (int oc = 0; oc < NOC; ++oc) {
        const int ocBase = oc * 128;
        const int Ntc    = ((Dout - ocBase) < 128) ? ((Dout - ocBase + 15) / 16) : 8;
        v8f acc[8];
#pragma unroll
        for (int t = 0; t < 8; ++t)
#pragma unroll
            for (int r = 0; r < 8; ++r) acc[t][r] = 0.0f;

        for (int kc = 0; kc < NKC; ++kc) {
            __syncthreads();
            // Stage W chunk (128 out-rows x 64 k) fp32 -> bf16, zero-padded.
            {
                const int row = tid >> 1;
                const int c0  = (tid & 1) * 32;
                const int gr  = ocBase + row;
                const float* src = Wg + (size_t)gr * Din + kc * KCH + c0;
                unsigned short* dst = wch + row * XS + c0;
#pragma unroll
                for (int j = 0; j < 32; j += 4) {
                    const int gc = kc * KCH + c0 + j;
                    float4 v;
                    if (gr < Dout && gc + 3 < Din) {
                        v = *(const float4*)(src + j);
                    } else {
                        v.x = (gr < Dout && gc + 0 < Din) ? src[j + 0] : 0.0f;
                        v.y = (gr < Dout && gc + 1 < Din) ? src[j + 1] : 0.0f;
                        v.z = (gr < Dout && gc + 2 < Din) ? src[j + 2] : 0.0f;
                        v.w = (gr < Dout && gc + 3 < Din) ? src[j + 3] : 0.0f;
                    }
                    pack4(dst + j, v);
                }
                if (gr < Dout && kc + 1 < NKC)
                    __builtin_prefetch(src + KCH, 0, 0);
            }
            __syncthreads();
#pragma unroll
            for (int ks = 0; ks < KCH / 32; ++ks) {
                v16bf a = load_frag_a(in_lds, in_stride, wave * 16, kc * KCH + ks * 32, lane);
                v16bf bfr[8];
#pragma unroll
                for (int tn = 0; tn < 8; ++tn)
                    if (tn < Ntc)       // wave-uniform guard: scalar branch
                        bfr[tn] = load_frag_b(wch, XS, tn * 16, ks * 32, lane);
                SCHED_FENCE();
#pragma unroll
                for (int tn = 0; tn < 8; ++tn)
                    if (tn < Ntc)
                        acc[tn] = wmma_bf16(a, bfr[tn], acc[tn]);
                SCHED_FENCE();
            }
        }
        __syncthreads();
        // bias + tanh + store bf16
#pragma unroll
        for (int tn = 0; tn < 8; ++tn) {
            if (tn < Ntc) {
                const int n = ocBase + tn * 16 + (lane & 15);
                if (n < Dout) {
                    const float bv    = bg[n];
                    const int   rbase = wave * 16 + (lane >> 4) * 8;
#pragma unroll
                    for (int r = 0; r < 8; ++r)
                        out_lds[(rbase + r) * out_stride + n] = f2bf(tanhf(acc[tn][r] + bv));
                }
            }
        }
    }
    // Zero-pad columns up to the next multiple of 64 (next layer reads them).
    constexpr int PadTo = (Dout + 63) & ~63;
    if constexpr (PadTo > Dout) {
        constexpr int PadW = PadTo - Dout;
        for (int e = tid; e < 128 * PadW; e += 256) {
            const int row = e / PadW;
            const int c   = Dout + (e % PadW);
            out_lds[row * out_stride + c] = 0;
        }
    }
    __syncthreads();
}

__global__ __launch_bounds__(256)
void som_mlp(MlpParams p) {
    extern __shared__ char smem[];
    unsigned short* hA   = (unsigned short*)(smem);            // h0 / h2 / h4
    unsigned short* hB   = (unsigned short*)(smem + OFF_B);    // x-chunk, h1/h3/h5
    unsigned short* wch  = (unsigned short*)(smem + OFF_W);    // weight chunk
    int*            sidx = (int*)(smem + OFF_IDX);
    float*          sred = (float*)(smem + OFF_RED);

    const int bk   = blockIdx.x;
    const int tid  = threadIdx.x;
    const int wave = tid >> 5;
    const int lane = tid & 31;

    const float* ctx = p.ctx_base + (size_t)bk * 2 * LL * DD;
    const float* ent = ctx + (size_t)LL * DD;

    if (tid < LL) sidx[tid] = p.idx[(size_t)bk * LL + tid];
    __syncthreads();

    // ---------------- Layer 0: x(128x1536) @ W0^T -> h0(128x768), tanh -----
    {
        const float* W0 = p.W[0];
        const float* b0 = p.Bv[0];
        for (int oc = 0; oc < 6; ++oc) {
            v8f acc[8];
#pragma unroll
            for (int t = 0; t < 8; ++t)
#pragma unroll
                for (int r = 0; r < 8; ++r) acc[t][r] = 0.0f;

            for (int kc = 0; kc < 24; ++kc) {
                __syncthreads();
                const int row = tid >> 1;
                const int c0  = (tid & 1) * 32;
                {   // W0 chunk (rows 0..767, cols 0..1535 -> always in range)
                    const float* src = W0 + (size_t)(oc * 128 + row) * 1536 + kc * KCH + c0;
                    unsigned short* dst = wch + row * XS + c0;
#pragma unroll
                    for (int j = 0; j < 32; j += 4)
                        pack4(dst + j, *(const float4*)(src + j));
                }
                {   // x chunk: first 768 cols = ctx row, rest = ent[idx[row]]
                    const int gk = kc * KCH + c0;
                    const float* src = (gk < DD)
                        ? (ctx + (size_t)row * DD + gk)
                        : (ent + (size_t)sidx[row] * DD + (gk - DD));
                    unsigned short* dst = hB + row * XS + c0;   // xbuf = region B
#pragma unroll
                    for (int j = 0; j < 32; j += 4)
                        pack4(dst + j, *(const float4*)(src + j));
                    __builtin_prefetch(src + KCH, 0, 0);
                }
                __syncthreads();
#pragma unroll
                for (int ks = 0; ks < KCH / 32; ++ks) {
                    v16bf a = load_frag_a(hB, XS, wave * 16, ks * 32, lane);
                    v16bf bfr[8];
#pragma unroll
                    for (int tn = 0; tn < 8; ++tn)
                        bfr[tn] = load_frag_b(wch, XS, tn * 16, ks * 32, lane);
                    SCHED_FENCE();
#pragma unroll
                    for (int tn = 0; tn < 8; ++tn)
                        acc[tn] = wmma_bf16(a, bfr[tn], acc[tn]);
                    SCHED_FENCE();
                }
            }
            __syncthreads();
#pragma unroll
            for (int tn = 0; tn < 8; ++tn) {
                const int n     = oc * 128 + tn * 16 + (lane & 15);
                const float bv  = b0[n];
                const int rbase = wave * 16 + (lane >> 4) * 8;
#pragma unroll
                for (int r = 0; r < 8; ++r)
                    hA[(rbase + r) * S0 + n] = f2bf(tanhf(acc[tn][r] + bv));
            }
        }
        __syncthreads();
    }

    // ---------------- Layers 1..5 (ping-pong A<->B) -------------------------
    mlp_layer<768, 384>(hA, S0, hB, S1, wch, p.W[1], p.Bv[1], tid);
    mlp_layer<384, 192>(hB, S1, hA, S2, wch, p.W[2], p.Bv[2], tid);
    mlp_layer<192,  96>(hA, S2, hB, S3, wch, p.W[3], p.Bv[3], tid);
    mlp_layer< 96,  48>(hB, S3, hA, S4, wch, p.W[4], p.Bv[4], tid);
    mlp_layer< 48,  24>(hA, S4, hB, S5, wch, p.W[5], p.Bv[5], tid);

    // ---------------- Layer 6 (24 -> 1, no tanh) + row-sum ------------------
    {
        const float* W6 = p.W[6];
        const float  b6 = p.Bv[6][0];
        float part = 0.0f;
        if (tid < LL) {
            float s = b6;
#pragma unroll
            for (int j = 0; j < 24; ++j)
                s += bf2f(hB[tid * S5 + j]) * W6[j];
            part = s;
        }
        sred[tid] = part;
        __syncthreads();
        for (int off = 128; off > 0; off >>= 1) {
            if (tid < off) sred[tid] += sred[tid + off];
            __syncthreads();
        }
        if (tid == 0) p.out[bk] = sred[0];
    }
}

// ---------------------------------------------------------------------------
// Host launcher. Inputs (setup_inputs dict order, tuples flattened):
//   d_in[0]      = context  (B,K,2,L,D) f32
//   d_in[1..7]   = Ws[0..6] f32
//   d_in[8..14]  = bs[0..6] f32
// d_out = (B,K) f32. d_ws holds idx (B*K*L ints = 512 KB).
// ---------------------------------------------------------------------------
extern "C" void kernel_launch(void* const* d_in, const int* in_sizes, int n_in,
                              void* d_out, int out_size, void* d_ws, size_t ws_size,
                              hipStream_t stream) {
    (void)in_sizes; (void)n_in; (void)out_size; (void)ws_size;

    const float* context = (const float*)d_in[0];
    int* idx = (int*)d_ws;

    // Opt in to >64KB dynamic LDS (no-op if already allowed).
    hipFuncSetAttribute((const void*)som_sim_argmax,
                        hipFuncAttributeMaxDynamicSharedMemorySize, SMEM1);
    hipFuncSetAttribute((const void*)som_mlp,
                        hipFuncAttributeMaxDynamicSharedMemorySize, SMEM2);

    som_sim_argmax<<<dim3(BK), dim3(256), SMEM1, stream>>>(context, idx);

    MlpParams p;
    p.ctx_base = context;
    p.idx      = idx;
    for (int i = 0; i < 7; ++i) {
        p.W[i]  = (const float*)d_in[1 + i];
        p.Bv[i] = (const float*)d_in[8 + i];
    }
    p.out = (float*)d_out;

    som_mlp<<<dim3(BK), dim3(256), SMEM2, stream>>>(p);
}
